// ClassifyMolBond_18923625906920
// MI455X (gfx1250) — compile-verified, matrix operand order checked by reading, not run on hardware
//
#include <hip/hip_runtime.h>
#include <hip/hip_bf16.h>

// ---------------------------------------------------------------------------
// ClassifyMolBond — CDNA5 (gfx1250) implementation.
// MPN loop (100 steps of [E,64]@[64,64]) via v_wmma_f32_16x16x32_f16,
// MLP tail via a generic WMMA GEMM. f32 storage, f16 WMMA operands,
// f32 accumulation. Tile-uniform epilogue guards, h0 prefetch.
// ---------------------------------------------------------------------------

typedef float     v8f  __attribute__((ext_vector_type(8)));
typedef _Float16  v16h __attribute__((ext_vector_type(16)));
typedef _Float16  v8h  __attribute__((ext_vector_type(8)));

#define BOND_H 64
#define ATOM_H 32

__device__ __forceinline__ int kmap(int j, int g) {
    // 16-bit A/B fragment K mapping (ISA 7.12.2): lanes 0-15 hold K g*8+j (j<8)
    // and 16+g*8+(j-8); lanes 16-31 the other halves.
    return (j < 8) ? (g * 8 + j) : (16 + g * 8 + (j - 8));
}

__device__ __forceinline__ void atomAddF(float* p, float v) {
    __hip_atomic_fetch_add(p, v, __ATOMIC_RELAXED, __HIP_MEMORY_SCOPE_AGENT);
}

// ---------------------------------------------------------------------------
// 1. Build merged directed edge list (src, dst)
// ---------------------------------------------------------------------------
__global__ void k_build_edges(const int* __restrict__ ae, const int* __restrict__ be,
                              const int* __restrict__ pb,
                              int Na, int Ea, int Eb, int NC, int E,
                              int* __restrict__ src, int* __restrict__ dst) {
    int e = blockIdx.x * blockDim.x + threadIdx.x;
    if (e >= E) return;
    int s, d;
    if (e < Ea) {
        s = ae[2 * e];            d = ae[2 * e + 1];
    } else if (e < Ea + Eb) {
        int i = e - Ea;
        s = be[2 * i] + Na;       d = be[2 * i + 1] + Na;
    } else if (e < Ea + Eb + NC) {
        int i = e - (Ea + Eb);
        s = pb[i];                d = pb[NC + i] + Na;
    } else {
        int i = e - (Ea + Eb + NC);
        s = pb[NC + i] + Na;      d = pb[i];
    }
    src[e] = s; dst[e] = d;
}

// ---------------------------------------------------------------------------
// 2. h0 = relu([x[src], ef] @ W_in + b_in)   (input dim 6 -> 64; scalar)
// ---------------------------------------------------------------------------
__global__ void k_h0(const float* __restrict__ ax, const float* __restrict__ aef,
                     const float* __restrict__ bx, const float* __restrict__ bef,
                     const int* __restrict__ src,
                     int Na, int Ea, int Eb, int E,
                     const float* __restrict__ Win, const float* __restrict__ bin,
                     float* __restrict__ h0) {
    long gid = (long)blockIdx.x * blockDim.x + threadIdx.x;
    if (gid >= (long)E * BOND_H) return;
    int e = (int)(gid >> 6), j = (int)(gid & 63);
    int s = src[e];
    const float* xr = (s < Na) ? &ax[(long)s * 5] : &bx[(long)(s - Na) * 5];
    float in5 = (e < Ea) ? aef[e] : ((e < Ea + Eb) ? bef[e - Ea] : 999.0f);
    float acc = bin[j];
    #pragma unroll
    for (int k = 0; k < 5; ++k) acc += xr[k] * Win[k * BOND_H + j];
    acc += in5 * Win[5 * BOND_H + j];
    h0[gid] = fmaxf(acc, 0.0f);
}

// ---------------------------------------------------------------------------
// 3. Pre-swizzle W_msg (f32 [64,64], row=K col=N) into f16 WMMA B fragments:
//    frag f = kt*4+nt ; layout Wf[(f*32 + lane)*16 + j]
// ---------------------------------------------------------------------------
__global__ void k_conv_wmsg(const float* __restrict__ Wmsg, _Float16* __restrict__ Wf) {
    int tid = blockIdx.x * blockDim.x + threadIdx.x;
    if (tid >= 2 * 4 * 32 * 16) return;
    int j = tid & 15;
    int lane = (tid >> 4) & 31;
    int f = tid >> 9;
    int kt = f >> 2, nt = f & 3;
    int g = lane >> 4;
    int n = nt * 16 + (lane & 15);
    int k = kt * 32 + kmap(j, g);
    Wf[tid] = (_Float16)Wmsg[k * BOND_H + n];
}

// ---------------------------------------------------------------------------
// 4. zero buffer
// ---------------------------------------------------------------------------
__global__ void k_zero(float* __restrict__ p, long n) {
    long gid = (long)blockIdx.x * blockDim.x + threadIdx.x;
    if (gid < n) p[gid] = 0.0f;
}

// ---------------------------------------------------------------------------
// 5. scatter-add: agg[dst[e]] += h[e]   (float global atomics, float4 granules)
// ---------------------------------------------------------------------------
__global__ void k_scatter(const float* __restrict__ h, const int* __restrict__ dst,
                          float* __restrict__ agg, int E) {
    long gid = (long)blockIdx.x * blockDim.x + threadIdx.x;
    if (gid >= (long)E * 16) return;
    int e = (int)(gid >> 4);
    int q = (int)(gid & 15) * 4;
    int d = dst[e];
    const float4 v = *(const float4*)&h[(long)e * BOND_H + q];
    float* ap = &agg[(long)d * BOND_H + q];
    atomAddF(ap + 0, v.x);
    atomAddF(ap + 1, v.y);
    atomAddF(ap + 2, v.z);
    atomAddF(ap + 3, v.w);
}

// ---------------------------------------------------------------------------
// 6. Message GEMM: h_new[e] = relu(h0[e] + agg[src[e]] @ W_msg + b_msg)
//    One wave per 16-edge tile; 8 x v_wmma_f32_16x16x32_f16 per wave.
// ---------------------------------------------------------------------------
__global__ void k_msg_gemm(const float* __restrict__ agg, const int* __restrict__ src,
                           const float* __restrict__ h0, const _Float16* __restrict__ Wf,
                           const float* __restrict__ bmsg, float* __restrict__ hout,
                           int E) {
    __shared__ __align__(16) _Float16 aS[8][16][72];   // 72-half pitch: 144B, 16B multiple
    const int wave = threadIdx.x >> 5;
    const int lane = threadIdx.x & 31;
    const long tile = (long)blockIdx.x * 8 + wave;
    const int e0 = (int)(tile * 16);
    const bool full = (e0 + 16 <= E);

    // prefetch h0 tile (consumed only in the epilogue, after the WMMA chain)
    if (e0 < E) __builtin_prefetch(&h0[(long)e0 * BOND_H + lane * 32]);

    // stage 16 rows x 64 f32 of gathered agg, converted to f16
    if (full) {
        #pragma unroll
        for (int i = 0; i < 8; ++i) {
            int c = lane * 8 + i;            // 0..255
            int r = c >> 4;                  // row in tile
            int cc = (c & 15) * 4;           // column
            int s = src[e0 + r];
            float4 v = *(const float4*)&agg[(long)s * BOND_H + cc];
            aS[wave][r][cc + 0] = (_Float16)v.x;
            aS[wave][r][cc + 1] = (_Float16)v.y;
            aS[wave][r][cc + 2] = (_Float16)v.z;
            aS[wave][r][cc + 3] = (_Float16)v.w;
        }
    } else {
        #pragma unroll
        for (int i = 0; i < 8; ++i) {
            int c = lane * 8 + i;
            int r = c >> 4;
            int cc = (c & 15) * 4;
            int e = e0 + r;
            float4 v = make_float4(0.f, 0.f, 0.f, 0.f);
            if (e < E) {
                int s = src[e];
                v = *(const float4*)&agg[(long)s * BOND_H + cc];
            }
            aS[wave][r][cc + 0] = (_Float16)v.x;
            aS[wave][r][cc + 1] = (_Float16)v.y;
            aS[wave][r][cc + 2] = (_Float16)v.z;
            aS[wave][r][cc + 3] = (_Float16)v.w;
        }
    }
    __syncthreads();

    const int row = lane & 15;
    const int g   = lane >> 4;
    const _Float16* rp = aS[wave][row];
    v8h lo0 = *(const v8h*)&rp[g * 8];
    v8h hi0 = *(const v8h*)&rp[16 + g * 8];
    v8h lo1 = *(const v8h*)&rp[32 + g * 8];
    v8h hi1 = *(const v8h*)&rp[48 + g * 8];
    v16h a0 = __builtin_shufflevector(lo0, hi0, 0,1,2,3,4,5,6,7,8,9,10,11,12,13,14,15);
    v16h a1 = __builtin_shufflevector(lo1, hi1, 0,1,2,3,4,5,6,7,8,9,10,11,12,13,14,15);

    v8f acc[4];
    #pragma unroll
    for (int nt = 0; nt < 4; ++nt) {
        v8f z = {0.f,0.f,0.f,0.f,0.f,0.f,0.f,0.f};
        acc[nt] = z;
        v16h b0 = *(const v16h*)&Wf[((0 * 4 + nt) * 32 + lane) * 16];
        v16h b1 = *(const v16h*)&Wf[((1 * 4 + nt) * 32 + lane) * 16];
        acc[nt] = __builtin_amdgcn_wmma_f32_16x16x32_f16(false, a0, false, b0,
                                                         (short)0, acc[nt], false, false);
        acc[nt] = __builtin_amdgcn_wmma_f32_16x16x32_f16(false, a1, false, b1,
                                                         (short)0, acc[nt], false, false);
    }

    const int cl = lane & 15;
    if (full) {
        // unguarded epilogue: per-(nt,r8) a half-wave writes 64B contiguous
        #pragma unroll
        for (int nt = 0; nt < 4; ++nt) {
            int col = nt * 16 + cl;
            float bb = bmsg[col];
            #pragma unroll
            for (int r8 = 0; r8 < 8; ++r8) {
                long idx = (long)(e0 + g * 8 + r8) * BOND_H + col;
                hout[idx] = fmaxf(acc[nt][r8] + h0[idx] + bb, 0.0f);
            }
        }
    } else {
        #pragma unroll
        for (int nt = 0; nt < 4; ++nt) {
            int col = nt * 16 + cl;
            float bb = bmsg[col];
            #pragma unroll
            for (int r8 = 0; r8 < 8; ++r8) {
                int e = e0 + g * 8 + r8;
                if (e < E) {
                    long idx = (long)e * BOND_H + col;
                    hout[idx] = fmaxf(acc[nt][r8] + h0[idx] + bb, 0.0f);
                }
            }
        }
    }
}

// ---------------------------------------------------------------------------
// 7. node head: node_h = relu([x, agg] @ W_node + b_node)  (69 -> 32, scalar)
// ---------------------------------------------------------------------------
__global__ void k_node(const float* __restrict__ ax, const float* __restrict__ bx,
                       const float* __restrict__ agg,
                       const float* __restrict__ Wn, const float* __restrict__ bn,
                       int Na, int N, float* __restrict__ node_h) {
    long gid = (long)blockIdx.x * blockDim.x + threadIdx.x;
    if (gid >= (long)N * ATOM_H) return;
    int n = (int)(gid >> 5), j = (int)(gid & 31);
    const float* xr = (n < Na) ? &ax[(long)n * 5] : &bx[(long)(n - Na) * 5];
    float acc = bn[j];
    #pragma unroll
    for (int k = 0; k < 5; ++k) acc += xr[k] * Wn[k * ATOM_H + j];
    const float* ar = &agg[(long)n * BOND_H];
    for (int k = 0; k < BOND_H; ++k) acc += ar[k] * Wn[(5 + k) * ATOM_H + j];
    node_h[gid] = fmaxf(acc, 0.0f);
}

// ---------------------------------------------------------------------------
// 8. assemble MLP input: [node_h[pb0], node_h[Na+pb1], bond_h]   (128 wide)
// ---------------------------------------------------------------------------
__global__ void k_mlp_in(const float* __restrict__ node_h, const float* __restrict__ h,
                         const int* __restrict__ pb,
                         int Na, int NC, int off, float* __restrict__ out) {
    long gid = (long)blockIdx.x * blockDim.x + threadIdx.x;
    if (gid >= (long)NC * 128) return;
    int i = (int)(gid >> 7), j = (int)(gid & 127);
    float v;
    if (j < 32) {
        v = node_h[(long)pb[i] * ATOM_H + j];
    } else if (j < 64) {
        v = node_h[(long)(Na + pb[NC + i]) * ATOM_H + (j - 32)];
    } else {
        int c = j - 64;
        v = h[(long)(off + i) * BOND_H + c] + h[(long)(off + NC + i) * BOND_H + c];
    }
    out[gid] = v;
}

// ---------------------------------------------------------------------------
// 9. Generic WMMA GEMM: C = relu(A[M,K] @ W[K,N] + b)
//    4 waves/block, 16 rows per wave; A staged f32->f16 in dynamic LDS,
//    W fragments gathered+converted on the fly (weights are L2-resident).
//    Requires K % 32 == 0, N % 16 == 0.
// ---------------------------------------------------------------------------
__global__ void k_gemm_relu(const float* __restrict__ A, const float* __restrict__ W,
                            const float* __restrict__ bias, float* __restrict__ C,
                            int M, int K, int N, int doRelu) {
    extern __shared__ __align__(16) _Float16 smem[];
    const int wave = threadIdx.x >> 5;
    const int lane = threadIdx.x & 31;
    const int pitch = K + 8;                    // halves; 2*(K+8) bytes, 16B multiple
    _Float16* aS = smem + (long)wave * 16 * pitch;
    const int rowBase = (blockIdx.x * 4 + wave) * 16;
    const bool full = (rowBase + 16 <= M);

    // stage 16 x K activations, f32 -> f16
    const int cpr = K >> 2;                     // float4 chunks per row
    if (full) {
        for (int c = lane; c < 4 * K; c += 32) {
            int r = c / cpr;
            int cc = (c % cpr) * 4;
            float4 v = *(const float4*)&A[(long)(rowBase + r) * K + cc];
            _Float16* dp = &aS[r * pitch + cc];
            dp[0] = (_Float16)v.x; dp[1] = (_Float16)v.y;
            dp[2] = (_Float16)v.z; dp[3] = (_Float16)v.w;
        }
    } else {
        for (int c = lane; c < 4 * K; c += 32) {
            int r = c / cpr;
            int cc = (c % cpr) * 4;
            int row = rowBase + r;
            float4 v = make_float4(0.f, 0.f, 0.f, 0.f);
            if (row < M) v = *(const float4*)&A[(long)row * K + cc];
            _Float16* dp = &aS[r * pitch + cc];
            dp[0] = (_Float16)v.x; dp[1] = (_Float16)v.y;
            dp[2] = (_Float16)v.z; dp[3] = (_Float16)v.w;
        }
    }
    __syncthreads();

    const int g  = lane >> 4;
    const int cl = lane & 15;
    const int NT = N >> 4, KT = K >> 5;

    for (int nt = 0; nt < NT; ++nt) {
        v8f acc = {0.f,0.f,0.f,0.f,0.f,0.f,0.f,0.f};
        int n = nt * 16 + cl;
        for (int kt = 0; kt < KT; ++kt) {
            const _Float16* rp = &aS[(lane & 15) * pitch + kt * 32];
            v8h lo = *(const v8h*)&rp[g * 8];
            v8h hi = *(const v8h*)&rp[16 + g * 8];
            v16h a = __builtin_shufflevector(lo, hi, 0,1,2,3,4,5,6,7,8,9,10,11,12,13,14,15);
            v16h bw;
            #pragma unroll
            for (int j = 0; j < 16; ++j) {
                int k = kt * 32 + kmap(j, g);
                bw[j] = (_Float16)W[(long)k * N + n];
            }
            acc = __builtin_amdgcn_wmma_f32_16x16x32_f16(false, a, false, bw,
                                                         (short)0, acc, false, false);
        }
        float bb = bias[n];
        if (full) {
            #pragma unroll
            for (int r8 = 0; r8 < 8; ++r8) {
                float v = acc[r8] + bb;
                if (doRelu) v = fmaxf(v, 0.0f);
                C[(long)(rowBase + g * 8 + r8) * N + n] = v;
            }
        } else {
            #pragma unroll
            for (int r8 = 0; r8 < 8; ++r8) {
                int row = rowBase + g * 8 + r8;
                if (row < M) {
                    float v = acc[r8] + bb;
                    if (doRelu) v = fmaxf(v, 0.0f);
                    C[(long)row * N + n] = v;
                }
            }
        }
    }
}

// ---------------------------------------------------------------------------
// 10. final layer (64 -> 4) + softmax + argmax + output packing
//     d_out layout: [NC,3] result (as floats) then [NC,4] probs
// ---------------------------------------------------------------------------
__global__ void k_final(const float* __restrict__ z5, const float* __restrict__ W6,
                        const float* __restrict__ b6, const int* __restrict__ pb,
                        int NC, float* __restrict__ out) {
    int i = blockIdx.x * blockDim.x + threadIdx.x;
    if (i >= NC) return;
    float lg[4] = {b6[0], b6[1], b6[2], b6[3]};
    const float* zr = &z5[(long)i * 64];
    for (int k = 0; k < 64; ++k) {
        float v = zr[k];
        #pragma unroll
        for (int c = 0; c < 4; ++c) lg[c] += v * W6[k * 4 + c];
    }
    float m = lg[0];
    #pragma unroll
    for (int c = 1; c < 4; ++c) m = fmaxf(m, lg[c]);
    float s = 0.f, ex[4];
    #pragma unroll
    for (int c = 0; c < 4; ++c) { ex[c] = __expf(lg[c] - m); s += ex[c]; }
    float inv = 1.0f / s;
    int am = 0; float best = lg[0];
    #pragma unroll
    for (int c = 1; c < 4; ++c) if (lg[c] > best) { best = lg[c]; am = c; }
    out[(long)i * 3 + 0] = (float)pb[i];
    out[(long)i * 3 + 1] = (float)pb[NC + i];
    out[(long)i * 3 + 2] = (float)am;
    float* pr = out + (long)NC * 3 + (long)i * 4;
    #pragma unroll
    for (int c = 0; c < 4; ++c) pr[c] = ex[c] * inv;
}

// ---------------------------------------------------------------------------
// host launcher
// ---------------------------------------------------------------------------
extern "C" void kernel_launch(void* const* d_in, const int* in_sizes, int n_in,
                              void* d_out, int out_size, void* d_ws, size_t ws_size,
                              hipStream_t stream) {
    (void)n_in; (void)out_size; (void)ws_size;
    const float* ax  = (const float*)d_in[0];
    const float* aef = (const float*)d_in[1];
    const int*   ae  = (const int*)  d_in[2];
    const float* bx  = (const float*)d_in[3];
    const float* bef = (const float*)d_in[4];
    const int*   be  = (const int*)  d_in[5];
    const int*   pb  = (const int*)  d_in[6];
    const float* Win = (const float*)d_in[7];
    const float* bin = (const float*)d_in[8];
    const float* Wmsg= (const float*)d_in[9];
    const float* bmsg= (const float*)d_in[10];
    const float* Wn  = (const float*)d_in[11];
    const float* bn  = (const float*)d_in[12];
    const float* Wl[6] = {(const float*)d_in[13], (const float*)d_in[15],
                          (const float*)d_in[17], (const float*)d_in[19],
                          (const float*)d_in[21], (const float*)d_in[23]};
    const float* bl[6] = {(const float*)d_in[14], (const float*)d_in[16],
                          (const float*)d_in[18], (const float*)d_in[20],
                          (const float*)d_in[22], (const float*)d_in[24]};

    const int Na = in_sizes[0] / 5;
    const int Ea = in_sizes[1];
    const int Nb = in_sizes[3] / 5;
    const int Eb = in_sizes[4];
    const int NC = in_sizes[6] / 2;
    const int N  = Na + Nb;
    const int E  = Ea + Eb + 2 * NC;

    // workspace bump allocator (256B aligned)
    char* w = (char*)d_ws;
    auto alloc = [&](size_t bytes) -> void* {
        void* p = (void*)w;
        w += (bytes + 255) & ~(size_t)255;
        return p;
    };
    int*       srcA  = (int*)      alloc((size_t)E * sizeof(int));
    int*       dstA  = (int*)      alloc((size_t)E * sizeof(int));
    float*     h0    = (float*)    alloc((size_t)E * BOND_H * sizeof(float));
    float*     hA    = (float*)    alloc((size_t)E * BOND_H * sizeof(float));
    float*     hB    = (float*)    alloc((size_t)E * BOND_H * sizeof(float));
    float*     agg   = (float*)    alloc((size_t)N * BOND_H * sizeof(float));
    _Float16*  Wf    = (_Float16*) alloc((size_t)4096 * sizeof(_Float16));
    float*     nodeH = (float*)    alloc((size_t)N * ATOM_H * sizeof(float));
    float*     zA    = (float*)    alloc((size_t)NC * 256 * sizeof(float));
    float*     zB    = (float*)    alloc((size_t)NC * 256 * sizeof(float));

    const int TB = 256;
    auto cdiv = [](long a, long b) { return (int)((a + b - 1) / b); };

    // graph construction + h0 + weight pre-swizzle
    k_build_edges<<<cdiv(E, TB), TB, 0, stream>>>(ae, be, pb, Na, Ea, Eb, NC, E, srcA, dstA);
    k_h0<<<cdiv((long)E * BOND_H, TB), TB, 0, stream>>>(ax, aef, bx, bef, srcA,
                                                        Na, Ea, Eb, E, Win, bin, h0);
    k_conv_wmsg<<<cdiv(4096, TB), TB, 0, stream>>>(Wmsg, Wf);

    // 100 message-passing steps
    const long aggN  = (long)N * BOND_H;
    const int  zeroG = cdiv(aggN, TB);
    const int  scatG = cdiv((long)E * 16, TB);
    const int  gemmG = cdiv((long)E, 16L * 8L);   // one wave per 16-edge tile, 8 waves/block
    const float* cur = h0;
    float* nxt = hA;
    for (int step = 0; step < 100; ++step) {
        k_zero<<<zeroG, TB, 0, stream>>>(agg, aggN);
        k_scatter<<<scatG, TB, 0, stream>>>(cur, dstA, agg, E);
        k_msg_gemm<<<gemmG, 256, 0, stream>>>(agg, srcA, h0, Wf, bmsg, nxt, E);
        cur = nxt;
        nxt = (nxt == hA) ? hB : hA;
    }

    // final aggregation + node head
    k_zero<<<zeroG, TB, 0, stream>>>(agg, aggN);
    k_scatter<<<scatG, TB, 0, stream>>>(cur, dstA, agg, E);
    k_node<<<cdiv((long)N * ATOM_H, TB), TB, 0, stream>>>(ax, bx, agg, Wn, bn, Na, N, nodeH);

    // MLP input assembly
    k_mlp_in<<<cdiv((long)NC * 128, TB), TB, 0, stream>>>(nodeH, cur, pb, Na, NC, Ea + Eb, zA);

    // MLP layers 1..5 via WMMA GEMM (4 waves/block, 64 rows/block)
    struct { const float* A; float* C; int K, Nn; int li; } layers[5] = {
        { zA, zB, 128, 128, 0 },
        { zB, zA, 128, 256, 1 },
        { zA, zB, 256, 256, 2 },
        { zB, zA, 256, 128, 3 },
        { zA, zB, 128,  64, 4 },
    };
    for (int l = 0; l < 5; ++l) {
        int K = layers[l].K, Nn = layers[l].Nn;
        size_t shm = (size_t)4 * 16 * (K + 8) * sizeof(_Float16);
        k_gemm_relu<<<cdiv(NC, 64), 128, shm, stream>>>(
            layers[l].A, Wl[layers[l].li], bl[layers[l].li], layers[l].C,
            NC, K, Nn, /*doRelu=*/1);
    }

    // final 64->4 + softmax + argmax + packing
    k_final<<<cdiv(NC, TB), TB, 0, stream>>>(zB, Wl[5], bl[5], pb, NC, (float*)d_out);
}
